// MultiHeadAttentionQuantum_65481071398175
// MI455X (gfx1250) — compile-verified, matrix operand order checked by reading, not run on hardware
//
#include <hip/hip_runtime.h>

// Shapes from the reference
#define B_   4
#define H_   16
#define S_   2048
#define Dk   64
#define E_   1024
#define KVP  88   // kv row pitch (halves): 176B rows -> 16B aligned, conflict-free
#define KVTP 40   // kvT row pitch (halves): 80B rows -> 16B aligned, conflict-free
#define PP   40   // P staging pitch (halves)
#define NIT  (S_ / 32)

// exp(s/8 - 8) == exp2(s * 0.125*log2e - 8*log2e); v_exp_f32 is exp2 natively
#define EXP_SCALE 0.18033688011112042f   // 0.125 * log2(e)
#define EXP_BIAS  (-11.541560327111708f) // -8 * log2(e)

typedef __attribute__((ext_vector_type(16))) _Float16 v16h;
typedef __attribute__((ext_vector_type(8)))  _Float16 v8h;
typedef __attribute__((ext_vector_type(8)))  float    v8f;

__device__ __forceinline__ float redsum16(float v) {
  #pragma unroll
  for (int m = 1; m < 16; m <<= 1) v += __shfl_xor(v, m, 32);
  return v;
}

// 8x cos(x + theta) on register data, packed to f16
__device__ __forceinline__ v8h cos8v(const float4 a, const float4 b,
                                     const float4 t, const float4 u) {
  v8h r;
  r[0] = (_Float16)__cosf(a.x + t.x); r[1] = (_Float16)__cosf(a.y + t.y);
  r[2] = (_Float16)__cosf(a.z + t.z); r[3] = (_Float16)__cosf(a.w + t.w);
  r[4] = (_Float16)__cosf(b.x + u.x); r[5] = (_Float16)__cosf(b.y + u.y);
  r[6] = (_Float16)__cosf(b.z + u.z); r[7] = (_Float16)__cosf(b.w + u.w);
  return r;
}

__device__ __forceinline__ v8h cos8(const float* __restrict__ xp,
                                    const float* __restrict__ tp) {
  return cos8v(*(const float4*)xp, *(const float4*)(xp + 4),
               *(const float4*)tp, *(const float4*)(tp + 4));
}

// Fragment = two contiguous 16B chunks at p and p+16 halves (A/B layout:
// grp0 holds K{0..7,16..23}, grp1 holds K{8..15,24..31}; caller adds grp*8)
__device__ __forceinline__ v16h frag_from_lds(const _Float16* p) {
  const v8h lo = *(const v8h*)p;
  const v8h hi = *(const v8h*)(p + 16);
  v16h r;
  #pragma unroll
  for (int j = 0; j < 8; ++j) { r[j] = lo[j]; r[8 + j] = hi[j]; }
  return r;
}

// -------------------------------------------------------------------------
// Fused flash attention, QKV = cos(x + theta) built on the fly.
// Fixed softmax shift (|score| <= 8) -> no running max, no in-loop shuffles.
// Software pipeline: double-buffered LDS KV tiles, one barrier per 32 keys;
// x tile for step i+1 staged in registers while step i's WMMAs execute.
// grid = (B*H)*(S/128), 256 threads (8 waves), 16 Q rows per wave.
// -------------------------------------------------------------------------
__global__ __launch_bounds__(256)
void qattn_flash(const float* __restrict__ x, const float* __restrict__ theta,
                 _Float16* __restrict__ att) {
  __shared__ _Float16 kv [2 * 32 * KVP];    // key rows (row-major, K-frags)
  __shared__ _Float16 kvT[2 * 64 * KVTP];   // transposed copy (V-frags)
  __shared__ _Float16 pb [8 * 16 * PP];     // per-wave P staging (16x32)

  const int bh = blockIdx.x >> 4, mt = blockIdx.x & 15;
  const int b = bh >> 4, h = bh & 15;
  const int wave = threadIdx.x >> 5, lane = threadIdx.x & 31;
  const int lm = lane & 15, grp = lane >> 4;

  const float* th = theta + h * Dk;
  const size_t xhead = (size_t)b * S_ * E_ + (size_t)h * Dk;
  const int mrow_base = mt * 128 + wave * 16;

  // ---- Q A-fragments (row = mrow_base + lm), k 0..31 / 32..63 ----
  v16h a0, a1;
  {
    const float* xr = x + xhead + (size_t)(mrow_base + lm) * E_;
    const v8h q0 = cos8(xr + grp * 8,      th + grp * 8);
    const v8h q1 = cos8(xr + 16 + grp * 8, th + 16 + grp * 8);
    const v8h q2 = cos8(xr + 32 + grp * 8, th + 32 + grp * 8);
    const v8h q3 = cos8(xr + 48 + grp * 8, th + 48 + grp * 8);
    #pragma unroll
    for (int j = 0; j < 8; ++j) {
      a0[j] = q0[j]; a0[8 + j] = q1[j];
      a1[j] = q2[j]; a1[8 + j] = q3[j];
    }
  }

  v8f o0 = {}, o1 = {}, o2 = {}, o3 = {};
  float srow[8];
  #pragma unroll
  for (int r = 0; r < 8; ++r) srow[r] = 0.0f;

  _Float16* pw = pb + wave * 16 * PP;
  const int br = threadIdx.x >> 3;        // build row 0..31
  const int bc = (threadIdx.x & 7) << 3;  // build col 0,8,...,56
  const float* xcol = x + xhead + bc;     // + s*E_
  const float4 tb0 = *(const float4*)(th + bc);      // loop-invariant theta
  const float4 tb1 = *(const float4*)(th + bc + 4);

  // ---- prologue: build tile 0 into buffer 0 ----
  {
    const float* src = xcol + (size_t)br * E_;
    const v8h cv = cos8v(*(const float4*)src, *(const float4*)(src + 4), tb0, tb1);
    *(v8h*)(kv + br * KVP + bc) = cv;
    #pragma unroll
    for (int j = 0; j < 8; ++j) kvT[(bc + j) * KVTP + br] = cv[j];
  }
  __syncthreads();

  for (int it = 0; it < NIT; ++it) {
    const int cur = it & 1;
    const _Float16* kvc  = kv  + cur * 32 * KVP;
    const _Float16* kvTc = kvT + cur * 64 * KVTP;
    _Float16* kvn  = kv  + (cur ^ 1) * 32 * KVP;
    _Float16* kvTn = kvT + (cur ^ 1) * 64 * KVTP;

    // ---- issue next tile's global loads immediately after the barrier ----
    float4 nxa, nxb;
    const bool more = (it + 1 < NIT);
    if (more) {
      const float* src = xcol + (size_t)((it + 1) * 32 + br) * E_;
      nxa = *(const float4*)src;
      nxb = *(const float4*)(src + 4);
      if (it + 2 < NIT)
        __builtin_prefetch(xcol + (size_t)((it + 2) * 32 + br) * E_, 0, 3);
    }

    // ---- scores: two 16-col tiles, K=64 as 2x WMMA K=32 ----
    v8f clo = {}, chi = {};
    {
      const _Float16* kr0 = kvc + lm * KVP;
      const v16h b0 = frag_from_lds(kr0 + grp * 8);
      const v16h b1 = frag_from_lds(kr0 + 32 + grp * 8);
      clo = __builtin_amdgcn_wmma_f32_16x16x32_f16(false, a0, false, b0, (short)0, clo, false, false);
      clo = __builtin_amdgcn_wmma_f32_16x16x32_f16(false, a1, false, b1, (short)0, clo, false, false);
      const _Float16* kr1 = kvc + (16 + lm) * KVP;
      const v16h b2 = frag_from_lds(kr1 + grp * 8);
      const v16h b3 = frag_from_lds(kr1 + 32 + grp * 8);
      chi = __builtin_amdgcn_wmma_f32_16x16x32_f16(false, a0, false, b2, (short)0, chi, false, false);
      chi = __builtin_amdgcn_wmma_f32_16x16x32_f16(false, a1, false, b3, (short)0, chi, false, false);
    }

    // ---- p = exp2(score*0.125*log2e - 8*log2e); lane-local row sums ----
    #pragma unroll
    for (int r = 0; r < 8; ++r) {
      const float pl = __builtin_amdgcn_exp2f(__fmaf_rn(clo[r], EXP_SCALE, EXP_BIAS));
      const float ph = __builtin_amdgcn_exp2f(__fmaf_rn(chi[r], EXP_SCALE, EXP_BIAS));
      srow[r] += pl + ph;
      // C/D layout: this lane holds row r+8*grp, col lm
      pw[(r + 8 * grp) * PP + lm]      = (_Float16)pl;
      pw[(r + 8 * grp) * PP + 16 + lm] = (_Float16)ph;
    }

    // ---- P as A-fragment (same-wave LDS ops are in-order) ----
    const v16h pa = frag_from_lds(pw + lm * PP + grp * 8);

    // ---- O += P(16x32) x V(32x16) per d-chunk; V-frags from kvT ----
    v16h vb;
    vb = frag_from_lds(kvTc + (0 * 16 + lm) * KVTP + grp * 8);
    o0 = __builtin_amdgcn_wmma_f32_16x16x32_f16(false, pa, false, vb, (short)0, o0, false, false);
    vb = frag_from_lds(kvTc + (1 * 16 + lm) * KVTP + grp * 8);
    o1 = __builtin_amdgcn_wmma_f32_16x16x32_f16(false, pa, false, vb, (short)0, o1, false, false);
    vb = frag_from_lds(kvTc + (2 * 16 + lm) * KVTP + grp * 8);
    o2 = __builtin_amdgcn_wmma_f32_16x16x32_f16(false, pa, false, vb, (short)0, o2, false, false);
    vb = frag_from_lds(kvTc + (3 * 16 + lm) * KVTP + grp * 8);
    o3 = __builtin_amdgcn_wmma_f32_16x16x32_f16(false, pa, false, vb, (short)0, o3, false, false);

    // ---- build next tile into the other buffer (pure VALU + ds_store) ----
    if (more) {
      const v8h cv = cos8v(nxa, nxb, tb0, tb1);
      *(v8h*)(kvn + br * KVP + bc) = cv;
      #pragma unroll
      for (int j = 0; j < 8; ++j) kvTn[(bc + j) * KVTP + br] = cv[j];
    }
    __syncthreads();
  }

  // ---- single end-of-loop row-sum reduction, normalize, store f16 ----
  #pragma unroll
  for (int r = 0; r < 8; ++r) {
    const float inv = 1.0f / redsum16(srow[r]);
    const int s = mrow_base + r + 8 * grp;
    const size_t base = ((size_t)b * S_ + s) * E_ + (size_t)h * Dk;
    att[base + lm]      = (_Float16)(o0[r] * inv);
    att[base + 16 + lm] = (_Float16)(o1[r] * inv);
    att[base + 32 + lm] = (_Float16)(o2[r] * inv);
    att[base + 48 + lm] = (_Float16)(o3[r] * inv);
  }
}

// -------------------------------------------------------------------------
// w_out f32 -> f16 (layout preserved; GEMM B-fragments read contiguous rows)
// -------------------------------------------------------------------------
__global__ __launch_bounds__(256)
void wcvt(const float* __restrict__ w, _Float16* __restrict__ wh) {
  const int i = (blockIdx.x * 256 + threadIdx.x) * 4;
  const float4 v = *(const float4*)(w + i);
  wh[i]     = (_Float16)v.x;
  wh[i + 1] = (_Float16)v.y;
  wh[i + 2] = (_Float16)v.z;
  wh[i + 3] = (_Float16)v.w;
}

// -------------------------------------------------------------------------
// out = att @ w_out^T + b_out via WMMA; M=8192, N=K=1024.
// Wave tile 16(M)x64(N); W (2MB f16) + att (16.8MB f16) live in 192MB L2.
// -------------------------------------------------------------------------
__global__ __launch_bounds__(256)
void out_gemm(const _Float16* __restrict__ att, const _Float16* __restrict__ wh,
              const float* __restrict__ bias, float* __restrict__ out) {
  const int nt = blockIdx.x >> 6;      // 64-wide N tile
  const int mtb = blockIdx.x & 63;     // 128-wide M tile
  const int wave = threadIdx.x >> 5, lane = threadIdx.x & 31;
  const int lm = lane & 15, grp = lane >> 4;
  const int mrow = mtb * 128 + wave * 16 + lm;
  const int nbase = nt * 64;

  const _Float16* arow = att + (size_t)mrow * E_;
  v8f acc[4] = {};

  #pragma unroll 2
  for (int k0 = 0; k0 < E_; k0 += 32) {
    v16h a;
    {
      const v8h lo = *(const v8h*)(arow + k0 + grp * 8);
      const v8h hi = *(const v8h*)(arow + k0 + 16 + grp * 8);
      #pragma unroll
      for (int j = 0; j < 8; ++j) { a[j] = lo[j]; a[8 + j] = hi[j]; }
    }
    #pragma unroll
    for (int j4 = 0; j4 < 4; ++j4) {
      const _Float16* wr = wh + (size_t)(nbase + j4 * 16 + lm) * E_;
      v16h bf;
      const v8h lo = *(const v8h*)(wr + k0 + grp * 8);
      const v8h hi = *(const v8h*)(wr + k0 + 16 + grp * 8);
      #pragma unroll
      for (int j = 0; j < 8; ++j) { bf[j] = lo[j]; bf[8 + j] = hi[j]; }
      acc[j4] = __builtin_amdgcn_wmma_f32_16x16x32_f16(false, a, false, bf, (short)0, acc[j4], false, false);
    }
  }

  const int row_out_base = mtb * 128 + wave * 16;
  #pragma unroll
  for (int j4 = 0; j4 < 4; ++j4) {
    const int n = nbase + j4 * 16 + lm;
    const float bz = bias[n];
    #pragma unroll
    for (int r = 0; r < 8; ++r) {
      const int m = row_out_base + r + 8 * grp;
      out[(size_t)m * E_ + n] = acc[j4][r] + bz;
    }
  }
}

// -------------------------------------------------------------------------
// Launch. Inputs (x, theta, w_out, b_out) fp32; output fp32 [B,S,E].
// Workspace: att f16 (16.78 MB) @ 0, w f16 (2.10 MB) after it.
// -------------------------------------------------------------------------
extern "C" void kernel_launch(void* const* d_in, const int* in_sizes, int n_in,
                              void* d_out, int out_size, void* d_ws, size_t ws_size,
                              hipStream_t stream) {
  (void)in_sizes; (void)n_in; (void)out_size; (void)ws_size;
  const float* x     = (const float*)d_in[0];
  const float* theta = (const float*)d_in[1];
  const float* w_out = (const float*)d_in[2];
  const float* b_out = (const float*)d_in[3];
  float* out = (float*)d_out;

  _Float16* att = (_Float16*)d_ws;
  _Float16* wh  = (_Float16*)((char*)d_ws + (size_t)B_ * S_ * E_ * sizeof(_Float16));

  wcvt<<<(E_ * E_) / (256 * 4), 256, 0, stream>>>(w_out, wh);
  qattn_flash<<<B_ * H_ * (S_ / 128), 256, 0, stream>>>(x, theta, att);
  out_gemm<<<(B_ * S_ / 128) * (E_ / 64), 256, 0, stream>>>(att, wh, b_out, out);
}
